// TestFP8_GEMM_41996190221084
// MI455X (gfx1250) — 8-bit matrix operand order repaired and checked by reading, compiled, not run on hardware
//
#include <hip/hip_runtime.h>


typedef __attribute__((ext_vector_type(16))) int   v16i;
typedef __attribute__((ext_vector_type(4)))  int   v4i;
typedef __attribute__((ext_vector_type(8)))  float v8f;
typedef __attribute__((ext_vector_type(4)))  float v4f;

static constexpr int GM = 32768;
static constexpr int GK = 4096;
static constexpr int GN = 128;
static constexpr int KCHUNK = 128;
static constexpr int NCHUNK = GK / KCHUNK;
static constexpr int WS_DWORDS = NCHUNK * 8 * 4 * 128;

#if __has_builtin(__builtin_amdgcn_cvt_pk_fp8_f32)
#define HAVE_HW_FP8 1
#endif

#ifndef HAVE_HW_FP8
__device__ inline unsigned sw_e4m3(float f) {
  unsigned u = __float_as_uint(f);
  unsigned s = (u >> 24) & 0x80u;
  unsigned a = u & 0x7FFFFFFFu;
  if (a >= 0x7F800000u) return s | 0x7Fu;
  unsigned lsb = (a >> 20) & 1u;
  a += 0x7FFFFu + lsb;
  int e = (int)(a >> 23) - 127;
  if (e < -6) {
    float scl = fabsf(f) * 512.0f;
    int q = (int)rintf(scl);
    if (q > 8) q = 8;
    return s | (unsigned)q;
  }
  unsigned ee = (unsigned)(e + 7);
  if (ee >= 16u) return s | 0x7Eu;
  return s | (ee << 3) | ((a >> 20) & 7u);
}
#endif

__device__ inline unsigned pack4_fp8(v4f f) {
#ifdef HAVE_HW_FP8
  int lo = __builtin_amdgcn_cvt_pk_fp8_f32(f.x, f.y, 0, false);
  return (unsigned)__builtin_amdgcn_cvt_pk_fp8_f32(f.z, f.w, lo, true);
#else
  return sw_e4m3(f.x) | (sw_e4m3(f.y) << 8) | (sw_e4m3(f.z) << 16) |
         (sw_e4m3(f.w) << 24);
#endif
}

__global__ __launch_bounds__(256) void pack_weight_fp8(
    const float* __restrict__ w, unsigned* __restrict__ wsB) {
  int tid = blockIdx.x * 256 + threadIdx.x;
  int j2   = tid & 3;
  int lane = (tid >> 2) & 31;
  int jq   = (tid >> 7) & 3;
  int t    = (tid >> 9) & 7;
  int c    = tid >> 12;
  int half = lane >> 4;
  int n    = t * 16 + (lane & 15);
  int k    = c * KCHUNK + jq * 32 + half * 16 + j2 * 4;
  v4f f = *(const v4f*)(w + (size_t)n * GK + k);
  wsB[tid] = pack4_fp8(f);
}

__device__ inline v16i load_bfrag(const unsigned* __restrict__ bc, int t) {
  v16i b;
#pragma unroll
  for (int jq = 0; jq < 4; ++jq) {
    v4i q = *(const v4i*)(bc + t * 512 + jq * 128);
    b[jq * 4 + 0] = q.x;
    b[jq * 4 + 1] = q.y;
    b[jq * 4 + 2] = q.z;
    b[jq * 4 + 3] = q.w;
  }
  return b;
}

__global__ __launch_bounds__(256) void fp8_gemm_16x128(
    const float* __restrict__ A, const unsigned* __restrict__ Bp,
    float* __restrict__ Out) {
  const int lane = threadIdx.x & 31;
  const int wave = threadIdx.x >> 5;
  const int half = lane >> 4;
  const int l15  = lane & 15;
  const long rowBase = ((long)blockIdx.x * 8 + wave) * 16;

  const float* arow = A + (rowBase + l15) * (long)GK + half * 16;

  v8f acc[8];
#pragma unroll
  for (int t = 0; t < 8; ++t) acc[t] = {};

  for (int c = 0; c < NCHUNK; ++c) {
    const float* ab = arow + c * KCHUNK;
    v16i a;
#pragma unroll
    for (int j = 0; j < 16; ++j) {
      const int k0 = ((j >> 2) << 5) + ((j & 3) << 2);
      v4f f = __builtin_nontemporal_load((const v4f*)(ab + k0));
      a[j] = (int)pack4_fp8(f);
    }

    const unsigned* bc = Bp + c * 4096 + lane * 4;
    v16i bcur = load_bfrag(bc, 0);
#pragma unroll
    for (int t = 0; t < 8; ++t) {
      v16i bnext;
      if (t < 7) bnext = load_bfrag(bc, t + 1);
      acc[t] = __builtin_amdgcn_wmma_f32_16x16x128_fp8_fp8(
          a, bcur, (short)0, acc[t], false, false);
      if (t < 7) bcur = bnext;
    }
  }

  float* outBase = Out + (rowBase + half * 8) * (long)GN + l15;
#pragma unroll
  for (int t = 0; t < 8; ++t) {
#pragma unroll
    for (int r = 0; r < 8; ++r) {
      __builtin_nontemporal_store(acc[t][r], outBase + (long)r * GN + t * 16);
    }
  }
}

extern "C" void kernel_launch(void* const* d_in, const int* in_sizes, int n_in,
                              void* d_out, int out_size, void* d_ws,
                              size_t ws_size, hipStream_t stream) {
  (void)in_sizes; (void)n_in; (void)out_size; (void)ws_size;
  const float* inp = (const float*)d_in[0];
  const float* wgt = (const float*)d_in[1];
  float* out = (float*)d_out;
  unsigned* wsB = (unsigned*)d_ws;

  pack_weight_fp8<<<WS_DWORDS / 256, 256, 0, stream>>>(wgt, wsB);

  fp8_gemm_16x128<<<GM / 128, 256, 0, stream>>>(inp, wsB, out);
}
